// Block_3659312136625
// MI455X (gfx1250) — matrix operand order repaired and checked by reading, not compiled, not run on hardware
//
// RWKV-7 block for MI455X (gfx1250, wave32, WMMA), v2.
//
// Key changes vs v1 (driven by disasm: scalarized guarded tile loads with
// per-element s_wait_loadcnt):
//  * all GEMM operands pre-converted to bf16 (weights once into ws, activation
//    producers emit bf16) -> tile loads are unguarded global_load_b128.
//  * 128x128 block tile, 8 waves x (2x4) 16x16x32 WMMA tiles, register
//    prefetch pipeline over K.
//  * r=16 bottleneck weights zero-padded to 32 so every GEMM has K%32==0,
//    M%128==0, N%8==0 -> no scalar fallback path.
//  * maa einsum as 4 strided WMMA GEMMs instead of per-token re-reads.
//  * wkv7 scan: 4 lanes per state row + shfl_xor reductions.
//
// Requires ws_size >= ~1.4 GB.

#include <hip/hip_runtime.h>

#define DEVFN __device__ __forceinline__

typedef __attribute__((ext_vector_type(16))) __bf16 bf16x16;
typedef __attribute__((ext_vector_type(8)))  float  floatx8;
typedef __attribute__((ext_vector_type(8)))  unsigned short ushortx8;

constexpr int CDIM = 2048;
constexpr int HSZ  = 64;
constexpr int NH   = CDIM / HSZ;   // 32 heads
constexpr int FFND = 8192;
constexpr int BB   = 4;
constexpr int TTD  = 2048;
constexpr int BT   = BB * TTD;     // 8192 tokens

DEVFN float bf16r(float x) { return (float)(__bf16)x; }
DEVFN ushortx8 ld8(const __bf16* p) { return *(const ushortx8*)p; }

// ---------------------------------------------------------------------------
// WMMA GEMM: Out[M,N] (+)= epi(A[M,K] @ B[K,N]); A,B bf16, f32 accum.
// 256 threads = 8 waves; block tile 128x128; wave tile 32x64 (2x4 WMMA).
// Assumes M%128==0, K%32==0, N%8==0 (guaranteed by padding in the driver).
// ---------------------------------------------------------------------------
enum { EPI_NONE = 0, EPI_TANH = 1, EPI_RELU2 = 2, EPI_SIGB = 3 };

template <typename OT, int EPI, bool ACC>
__global__ __launch_bounds__(256) void gemm_bf16(
    const __bf16* __restrict__ A, int lda,
    const __bf16* __restrict__ B, int ldb,
    OT* __restrict__ Out, const float* __restrict__ bias,
    int M, int N, int K) {
  constexpr int LDT = 40;                    // bf16 stride (80 B, 16B-aligned)
  __shared__ __bf16 As[128 * LDT];
  __shared__ __bf16 Bs[128 * LDT];
  const int tid  = threadIdx.x;
  const int lane = tid & 31;
  const int wave = tid >> 5;
  const int wm = wave >> 1, wn = wave & 1;   // wave grid 4(M) x 2(N)
  const int m0 = blockIdx.y * 128, n0 = blockIdx.x * 128;
  const int hh = lane >> 4, l16 = lane & 15;

  // tile-load thread mapping
  const int arow = tid >> 1, akh = (tid & 1) * 16;          // A: 128 x 32
  const int bkp = tid >> 4, bng = (tid & 15) * 8;           // B: 16 k-pairs x 128

  floatx8 acc[8];
#pragma unroll
  for (int i = 0; i < 2; ++i)
#pragma unroll
    for (int j = 0; j < 4; ++j)
#pragma unroll
      for (int r = 0; r < 8; ++r) {
        float c0 = 0.f;
        if (ACC) {
          int gm = m0 + wm * 32 + i * 16 + hh * 8 + r;
          int gn = n0 + wn * 64 + j * 16 + l16;
          if (gm < M && gn < N) c0 = (float)Out[(size_t)gm * N + gn];
        }
        acc[i * 4 + j][r] = c0;
      }

  ushortx8 ra0, ra1, rb0, rb1;
  auto loadTile = [&](int k0) {
    const __bf16* ap = A + (size_t)(m0 + arow) * lda + k0 + akh;
    ra0 = ld8(ap);
    ra1 = ld8(ap + 8);
    const ushortx8 z = {0, 0, 0, 0, 0, 0, 0, 0};
    if (n0 + bng < N) {
      const __bf16* bp = B + (size_t)(k0 + 2 * bkp) * ldb + n0 + bng;
      rb0 = ld8(bp);
      rb1 = ld8(bp + ldb);
    } else {
      rb0 = z;
      rb1 = z;
    }
  };
  auto storeTile = [&]() {
    *(ushortx8*)&As[arow * LDT + akh] = ra0;
    *(ushortx8*)&As[arow * LDT + akh + 8] = ra1;
    unsigned int* bs32 = (unsigned int*)Bs;
#pragma unroll
    for (int j = 0; j < 8; ++j) {
      unsigned int pk = (unsigned int)rb0[j] | ((unsigned int)rb1[j] << 16);
      bs32[(bng + j) * (LDT / 2) + bkp] = pk;  // Bs[n][k-pair]
    }
  };

  loadTile(0);
  for (int k0 = 0; k0 < K; k0 += 32) {
    storeTile();
    __syncthreads();
    if (k0 + 32 < K) loadTile(k0 + 32);  // overlap next loads with WMMA

    bf16x16 af[2], bfm[4];
#pragma unroll
    for (int i = 0; i < 2; ++i) {
      const int ml = wm * 32 + i * 16 + l16;
#pragma unroll
      for (int vg = 0; vg < 8; ++vg) {
        int kb = ((vg & 4) ? 16 : 0) + hh * 8 + (vg & 3) * 2;
        af[i][2 * vg]     = As[ml * LDT + kb];
        af[i][2 * vg + 1] = As[ml * LDT + kb + 1];
      }
    }
#pragma unroll
    for (int j = 0; j < 4; ++j) {
      const int nl = wn * 64 + j * 16 + l16;
#pragma unroll
      for (int vg = 0; vg < 8; ++vg) {
        int kb = ((vg & 4) ? 16 : 0) + hh * 8 + (vg & 3) * 2;
        bfm[j][2 * vg]     = Bs[nl * LDT + kb];
        bfm[j][2 * vg + 1] = Bs[nl * LDT + kb + 1];
      }
    }
#pragma unroll
    for (int i = 0; i < 2; ++i)
#pragma unroll
      for (int j = 0; j < 4; ++j)
        acc[i * 4 + j] = __builtin_amdgcn_wmma_f32_16x16x32_bf16(
            false, af[i], false, bfm[j], (short)0, acc[i * 4 + j], false, false);
    __syncthreads();
  }

#pragma unroll
  for (int i = 0; i < 2; ++i)
#pragma unroll
    for (int j = 0; j < 4; ++j)
#pragma unroll
      for (int r = 0; r < 8; ++r) {
        int gm = m0 + wm * 32 + i * 16 + hh * 8 + r;
        int gn = n0 + wn * 64 + j * 16 + l16;
        if (gm < M && gn < N) {
          float v = acc[i * 4 + j][r];
          if (EPI == EPI_TANH) v = tanhf(v);
          else if (EPI == EPI_RELU2) { v = fmaxf(v, 0.f); v = v * v; }
          else if (EPI == EPI_SIGB) {
            float bb = bias ? bias[gn] : 0.f;
            v = 1.f / (1.f + expf(-(v + bb)));
          }
          Out[(size_t)gm * N + gn] = (OT)v;
        }
      }
}

// ---------------------------------------------------------------------------
// converts (run once per launch; weights -> bf16 in ws)
// ---------------------------------------------------------------------------
__global__ __launch_bounds__(256) void cvt_kernel(
    const float* __restrict__ in, __bf16* __restrict__ out, size_t n) {
  size_t i = (size_t)blockIdx.x * 256 + threadIdx.x;
  if (i < n) out[i] = (__bf16)in[i];
}
// pad columns: in [rows,cin] -> out [rows,cout], extra cols zero
__global__ __launch_bounds__(256) void cvtpad2d_kernel(
    const float* __restrict__ in, __bf16* __restrict__ out,
    int rows, int cin, int cout) {
  size_t i = (size_t)blockIdx.x * 256 + threadIdx.x;
  size_t n = (size_t)rows * cout;
  if (i >= n) return;
  int r = (int)(i / cout), c = (int)(i % cout);
  out[i] = (c < cin) ? (__bf16)in[(size_t)r * cin + c] : (__bf16)0.f;
}
// linear pad: copy first nin elems, zero up to nout (pads rows of [16,C]->[32,C])
__global__ __launch_bounds__(256) void cvtpadlin_kernel(
    const float* __restrict__ in, __bf16* __restrict__ out,
    size_t nin, size_t nout) {
  size_t i = (size_t)blockIdx.x * 256 + threadIdx.x;
  if (i < nout) out[i] = (i < nin) ? (__bf16)in[i] : (__bf16)0.f;
}

// ---------------------------------------------------------------------------
// LayerNorm over C=2048 per token. Block = 256 threads = 1 token.
// ---------------------------------------------------------------------------
__global__ __launch_bounds__(256) void ln_kernel(
    const float* __restrict__ x, const float* __restrict__ w,
    const float* __restrict__ b, float* __restrict__ out) {
  const int tok = blockIdx.x;
  const float* row = x + (size_t)tok * CDIM;
  float* orow = out + (size_t)tok * CDIM;
  __shared__ float red[256];
  const int tid = threadIdx.x;
  float v[8];
  float s = 0.f;
#pragma unroll
  for (int e = 0; e < 8; ++e) { v[e] = row[tid + e * 256]; s += v[e]; }
  red[tid] = s; __syncthreads();
  for (int st = 128; st > 0; st >>= 1) {
    if (tid < st) red[tid] += red[tid + st];
    __syncthreads();
  }
  const float mu = red[0] * (1.f / CDIM);
  __syncthreads();
  float sv = 0.f;
#pragma unroll
  for (int e = 0; e < 8; ++e) { float d = v[e] - mu; sv += d * d; }
  red[tid] = sv; __syncthreads();
  for (int st = 128; st > 0; st >>= 1) {
    if (tid < st) red[tid] += red[tid + st];
    __syncthreads();
  }
  const float rs = rsqrtf(red[0] * (1.f / CDIM) + 1e-5f);
#pragma unroll
  for (int e = 0; e < 8; ++e) {
    int c = tid + e * 256;
    orow[c] = (v[e] - mu) * rs * w[c] + b[c];
  }
}

// xx = shift(in)-in (f32) ; mix = bf16(in + xx*coef)
__global__ __launch_bounds__(256) void shiftmix_kernel(
    const float* __restrict__ in, const float* __restrict__ coef,
    float* __restrict__ xxo, __bf16* __restrict__ mixo) {
  size_t idx = (size_t)blockIdx.x * 256 + threadIdx.x;
  int c = (int)(idx & (CDIM - 1));
  size_t tok = idx >> 11;
  int t = (int)(tok & (TTD - 1));
  float cur = in[idx];
  float prev = (t > 0) ? in[idx - CDIM] : 0.f;
  float d = prev - cur;
  xxo[idx] = d;
  mixo[idx] = (__bf16)(cur + d * coef[c]);
}

// out = bf16(in + (shift(in)-in)*coef)
__global__ __launch_bounds__(256) void shiftcoef_kernel(
    const float* __restrict__ in, const float* __restrict__ coef,
    __bf16* __restrict__ out) {
  size_t idx = (size_t)blockIdx.x * 256 + threadIdx.x;
  int c = (int)(idx & (CDIM - 1));
  size_t tok = idx >> 11;
  int t = (int)(tok & (TTD - 1));
  float cur = in[idx];
  float prev = (t > 0) ? in[idx - CDIM] : 0.f;
  out[idx] = (__bf16)(cur + (prev - cur) * coef[c]);
}

// xrg/xwa/xk/xv = bf16(xln + xx*(coef + m))
__global__ __launch_bounds__(256) void mix4_kernel(
    const float* __restrict__ xln, const float* __restrict__ xx,
    const float* __restrict__ mrg, const float* __restrict__ mwa,
    const float* __restrict__ mk_, const float* __restrict__ mv_,
    const float* __restrict__ crg, const float* __restrict__ cwa,
    const float* __restrict__ ck, const float* __restrict__ cv,
    __bf16* __restrict__ xrg, __bf16* __restrict__ xwa,
    __bf16* __restrict__ xk, __bf16* __restrict__ xv) {
  size_t idx = (size_t)blockIdx.x * 256 + threadIdx.x;
  int c = (int)(idx & (CDIM - 1));
  float cur = xln[idx], d = xx[idx];
  xrg[idx] = (__bf16)(cur + d * (crg[c] + mrg[idx]));
  xwa[idx] = (__bf16)(cur + d * (cwa[c] + mwa[idx]));
  xk[idx]  = (__bf16)(cur + d * (ck[c] + mk_[idx]));
  xv[idx]  = (__bf16)(cur + d * (cv[c] + mv_[idx]));
}

// wlog = -softplus(-(time_decay + wraw)) - 0.5   (in place)
__global__ __launch_bounds__(256) void decay_kernel(
    float* __restrict__ wlog, const float* __restrict__ td) {
  size_t idx = (size_t)blockIdx.x * 256 + threadIdx.x;
  int c = (int)(idx & (CDIM - 1));
  float y = td[c] + wlog[idx];
  float sny = -y;
  float spv = (sny > 20.f) ? sny : log1pf(expf(sny));
  wlog[idx] = -spv - 0.5f;
}

// k = (k*ma + k*a*(1-ma)) * exp(min(wlog*mk2, 0))   (in place)
__global__ __launch_bounds__(256) void kmod_kernel(
    float* __restrict__ k, const float* __restrict__ ma,
    const float* __restrict__ a, const float* __restrict__ wlog,
    const float* __restrict__ mk2) {
  size_t idx = (size_t)blockIdx.x * 256 + threadIdx.x;
  float kv = k[idx], mav = ma[idx], av = a[idx];
  kv = kv * mav + kv * av * (1.f - mav);
  float e = fminf(wlog[idx] * mk2[idx], 0.f);
  k[idx] = kv * expf(e);
}

DEVFN float blocksum64(float v, float* red, int tid) {
  __syncthreads();
  red[tid] = v; __syncthreads();
#pragma unroll
  for (int st = 32; st > 0; st >>= 1) {
    if (tid < st) red[tid] += red[tid + st];
    __syncthreads();
  }
  return red[0];
}

// kk = normalize_per_head(k + kkraw). Block = 64 threads = 1 (token, head).
__global__ __launch_bounds__(64) void kknorm_kernel(
    const float* __restrict__ k, const float* __restrict__ kkraw,
    float* __restrict__ kk) {
  const int th = blockIdx.x;
  const int tok = th >> 5, h = th & (NH - 1);
  const int tid = threadIdx.x;
  const size_t off = (size_t)tok * CDIM + h * HSZ + tid;
  __shared__ float red[64];
  float s = k[off] + kkraw[off];
  float ss = blocksum64(s * s, red, tid);
  kk[off] = s / fmaxf(sqrtf(ss), 1e-12f);
}

// wkv7 sequential scan. Block = 256 threads = one (b,h).
// 4 lanes per state row; each lane holds 16 state columns in registers.
__global__ __launch_bounds__(256) void wkv7_kernel(
    const float* __restrict__ r, const float* __restrict__ w,
    const float* __restrict__ k, const float* __restrict__ v,
    const float* __restrict__ kk, const float* __restrict__ asig,
    float* __restrict__ o) {
  const int bh = blockIdx.x;
  const int b = bh >> 5;
  const int h = bh & (NH - 1);
  const int tid = threadIdx.x;
  const int i = tid >> 2;       // state row 0..63
  const int p = tid & 3;        // column group (16 cols each)
  const int c0 = p * 16;
  __shared__ float rs[HSZ], ws[HSZ], ks[HSZ], vs[HSZ], as_[HSZ], bs_[HSZ];
  float S[16];
#pragma unroll
  for (int j = 0; j < 16; ++j) S[j] = 0.f;
  const size_t vecbase = ((size_t)b * TTD) * CDIM + h * HSZ;
  for (int t = 0; t < TTD; ++t) {
    const size_t off = vecbase + (size_t)t * CDIM;
    if (tid < HSZ) {
      float kkv = kk[off + tid];
      rs[tid] = bf16r(r[off + tid]);
      ws[tid] = expf(-expf(bf16r(w[off + tid])));
      ks[tid] = bf16r(k[off + tid]);
      vs[tid] = bf16r(v[off + tid]);
      as_[tid] = bf16r(-kkv);
      bs_[tid] = bf16r(kkv * asig[off + tid]);
    }
    __syncthreads();
    float Sa = 0.f;
#pragma unroll
    for (int j = 0; j < 16; ++j) Sa += S[j] * as_[c0 + j];
    Sa += __shfl_xor(Sa, 1, 32);
    Sa += __shfl_xor(Sa, 2, 32);
    const float vi = vs[i];
    float op = 0.f;
#pragma unroll
    for (int j = 0; j < 16; ++j) {
      float sj = S[j] * ws[c0 + j] + Sa * bs_[c0 + j] + vi * ks[c0 + j];
      S[j] = sj;
      op += sj * rs[c0 + j];
    }
    op += __shfl_xor(op, 1, 32);
    op += __shfl_xor(op, 2, 32);
    if (p == 0) o[off + i] = op;
    __syncthreads();
  }
}

// GroupNorm(o) + rk*v + gate -> bf16. Block = 64 threads = 1 (token, head).
__global__ __launch_bounds__(64) void gngate_kernel(
    const float* __restrict__ o, const float* __restrict__ r,
    const float* __restrict__ k, const float* __restrict__ v,
    const float* __restrict__ g, const float* __restrict__ faaaa,
    const float* __restrict__ gnw, const float* __restrict__ gnb,
    __bf16* __restrict__ og) {
  const int th = blockIdx.x;
  const int tok = th >> 5, h = th & (NH - 1);
  const int tid = threadIdx.x;
  const int c = h * HSZ + tid;
  const size_t off = (size_t)tok * CDIM + c;
  __shared__ float red[64];
  float ov = o[off];
  float mu = blocksum64(ov, red, tid) * (1.f / HSZ);
  float dv = ov - mu;
  float var = blocksum64(dv * dv, red, tid) * (1.f / HSZ);
  float on = dv * rsqrtf(var + 64e-5f) * gnw[c] + gnb[c];
  float rk = blocksum64(r[off] * k[off] * faaaa[c], red, tid);
  og[off] = (__bf16)((on + rk * v[off]) * g[off]);
}

__global__ __launch_bounds__(256) void add_kernel(
    const float* __restrict__ a, const float* __restrict__ b,
    float* __restrict__ out, size_t n) {
  size_t idx = (size_t)blockIdx.x * 256 + threadIdx.x;
  if (idx < n) out[idx] = a[idx] + b[idx];
}

// ---------------------------------------------------------------------------
extern "C" void kernel_launch(void* const* d_in, const int* in_sizes, int n_in,
                              void* d_out, int out_size, void* d_ws,
                              size_t ws_size, hipStream_t stream) {
  (void)in_sizes; (void)n_in; (void)out_size; (void)ws_size;
  const float* x        = (const float*)d_in[0];
  const float* ln1_w    = (const float*)d_in[1];
  const float* ln1_b    = (const float*)d_in[2];
  const float* ln2_w    = (const float*)d_in[3];
  const float* ln2_b    = (const float*)d_in[4];
  const float* maa_x    = (const float*)d_in[5];
  const float* maa_rg   = (const float*)d_in[6];
  const float* maa_wa   = (const float*)d_in[7];
  const float* maa_k    = (const float*)d_in[8];
  const float* maa_v    = (const float*)d_in[9];
  const float* time_dec = (const float*)d_in[10];
  const float* faaaa    = (const float*)d_in[11];
  const float* aaaaa    = (const float*)d_in[12];
  const float* maa_w1   = (const float*)d_in[13];
  const float* maa_w2   = (const float*)d_in[14];
  const float* dec_w1   = (const float*)d_in[15];
  const float* dec_w2   = (const float*)d_in[16];
  const float* aaa_w1   = (const float*)d_in[17];
  const float* aaa_w2   = (const float*)d_in[18];
  const float* kkk_w1   = (const float*)d_in[19];
  const float* kkk_w2   = (const float*)d_in[20];
  const float* gate_w1  = (const float*)d_in[21];
  const float* gate_w2  = (const float*)d_in[22];
  const float* ma_w1    = (const float*)d_in[23];
  const float* ma_w2    = (const float*)d_in[24];
  const float* misc_a   = (const float*)d_in[25];
  const float* mk_w1    = (const float*)d_in[26];
  const float* mk_w2    = (const float*)d_in[27];
  const float* misc_k   = (const float*)d_in[28];
  const float* W_r      = (const float*)d_in[29];
  const float* W_k      = (const float*)d_in[30];
  const float* W_v      = (const float*)d_in[31];
  const float* W_o      = (const float*)d_in[32];
  const float* gn_w     = (const float*)d_in[33];
  const float* gn_b     = (const float*)d_in[34];
  const float* cm_maa_k = (const float*)d_in[35];
  const float* cm_key   = (const float*)d_in[36];
  const float* cm_val   = (const float*)d_in[37];
  float* out = (float*)d_out;

  // ---- workspace layout (bump allocator) ---------------------------------
  const size_t SLOT = (size_t)BT * CDIM;
  float* fp = (float*)d_ws;
  auto allocF = [&](size_t n) { float* p = fp; fp += n; return p; };
  float* xln   = allocF(SLOT);  // later hln
  float* xx    = allocF(SLOT);
  float* mrg   = allocF(SLOT);  // later rbuf
  float* mwa   = allocF(SLOT);  // later wlog
  float* mk_m  = allocF(SLOT);  // later kbuf
  float* mv_m  = allocF(SLOT);  // later vbuf
  float* gbuf  = allocF(SLOT);
  float* kkbuf = allocF(SLOT);
  float* asig  = allocF(SLOT);
  float* mabuf = allocF(SLOT);
  float* mk2   = allocF(SLOT);
  float* kkraw = allocF(SLOT);
  float* obuf  = allocF(SLOT);
  float* oproj = allocF(SLOT);
  __bf16* bp = (__bf16*)fp;
  auto allocB = [&](size_t n) { __bf16* p = bp; bp += n; return p; };
  __bf16* xxx_b = allocB(SLOT);
  __bf16* xrg_b = allocB(SLOT);
  __bf16* xwa_b = allocB(SLOT);
  __bf16* xk_b  = allocB(SLOT);
  __bf16* xv_b  = allocB(SLOT);
  __bf16* og_b  = allocB(SLOT);
  __bf16* kf_b  = allocB(SLOT);
  __bf16* ffb   = allocB((size_t)BT * FFND);
  __bf16* m_t    = allocB((size_t)BT * 128);
  __bf16* gate_t = allocB((size_t)BT * 128);
  __bf16* dec_t  = allocB((size_t)BT * 64);
  __bf16* aaa_t  = allocB((size_t)BT * 32);
  __bf16* ma_t   = allocB((size_t)BT * 32);
  __bf16* mk_t   = allocB((size_t)BT * 32);
  __bf16* kkk_t  = allocB((size_t)BT * 32);
  __bf16* wb_r      = allocB((size_t)CDIM * CDIM);
  __bf16* wb_k      = allocB((size_t)CDIM * CDIM);
  __bf16* wb_v      = allocB((size_t)CDIM * CDIM);
  __bf16* wb_o      = allocB((size_t)CDIM * CDIM);
  __bf16* wb_cmkey  = allocB((size_t)CDIM * FFND);
  __bf16* wb_cmval  = allocB((size_t)FFND * CDIM);
  __bf16* wb_maa_w1 = allocB((size_t)CDIM * 128);
  __bf16* wb_maa_w2 = allocB((size_t)128 * CDIM);
  __bf16* wb_gate_w1 = allocB((size_t)CDIM * 128);
  __bf16* wb_gate_w2 = allocB((size_t)128 * CDIM);
  __bf16* wb_dec_w1  = allocB((size_t)CDIM * 64);
  __bf16* wb_dec_w2  = allocB((size_t)64 * CDIM);
  __bf16* wb_aaa_w1  = allocB((size_t)CDIM * 32);  // padded 16->32 cols
  __bf16* wb_aaa_w2  = allocB((size_t)32 * CDIM);  // padded 16->32 rows
  __bf16* wb_ma_w1   = allocB((size_t)CDIM * 32);
  __bf16* wb_ma_w2   = allocB((size_t)32 * CDIM);
  __bf16* wb_mk_w1   = allocB((size_t)CDIM * 32);
  __bf16* wb_mk_w2   = allocB((size_t)32 * CDIM);
  __bf16* wb_kkk_w1  = allocB((size_t)CDIM * 32);
  __bf16* wb_kkk_w2  = allocB((size_t)32 * CDIM);

  const size_t NELT = SLOT;
  const dim3 eltGrid((unsigned)(NELT / 256)), eltBlk(256);
  auto gg = [](int M, int N) { return dim3((N + 127) / 128, (M + 127) / 128); };
  auto cgrid = [](size_t n) { return dim3((unsigned)((n + 255) / 256)); };

  // ---- weight conversion to bf16 (+ K padding 16->32) --------------------
  auto cvt = [&](const float* in, __bf16* o2, size_t n) {
    cvt_kernel<<<cgrid(n), 256, 0, stream>>>(in, o2, n);
  };
  cvt(W_r, wb_r, (size_t)CDIM * CDIM);
  cvt(W_k, wb_k, (size_t)CDIM * CDIM);
  cvt(W_v, wb_v, (size_t)CDIM * CDIM);
  cvt(W_o, wb_o, (size_t)CDIM * CDIM);
  cvt(cm_key, wb_cmkey, (size_t)CDIM * FFND);
  cvt(cm_val, wb_cmval, (size_t)FFND * CDIM);
  cvt(maa_w1, wb_maa_w1, (size_t)CDIM * 128);
  cvt(maa_w2, wb_maa_w2, (size_t)128 * CDIM);
  cvt(gate_w1, wb_gate_w1, (size_t)CDIM * 128);
  cvt(gate_w2, wb_gate_w2, (size_t)128 * CDIM);
  cvt(dec_w1, wb_dec_w1, (size_t)CDIM * 64);
  cvt(dec_w2, wb_dec_w2, (size_t)64 * CDIM);
  auto padw1 = [&](const float* in, __bf16* o2) {
    cvtpad2d_kernel<<<cgrid((size_t)CDIM * 32), 256, 0, stream>>>(in, o2, CDIM, 16, 32);
  };
  auto padw2 = [&](const float* in, __bf16* o2) {
    cvtpadlin_kernel<<<cgrid((size_t)32 * CDIM), 256, 0, stream>>>(
        in, o2, (size_t)16 * CDIM, (size_t)32 * CDIM);
  };
  padw1(aaa_w1, wb_aaa_w1); padw2(aaa_w2, wb_aaa_w2);
  padw1(ma_w1, wb_ma_w1);   padw2(ma_w2, wb_ma_w2);
  padw1(mk_w1, wb_mk_w1);   padw2(mk_w2, wb_mk_w2);
  padw1(kkk_w1, wb_kkk_w1); padw2(kkk_w2, wb_kkk_w2);

  // ---- tmix --------------------------------------------------------------
  ln_kernel<<<BT, 256, 0, stream>>>(x, ln1_w, ln1_b, xln);
  shiftmix_kernel<<<eltGrid, eltBlk, 0, stream>>>(xln, maa_x, xx, xxx_b);

  gemm_bf16<__bf16, EPI_TANH, false><<<gg(BT, 128), 256, 0, stream>>>(
      xxx_b, CDIM, wb_maa_w1, 128, m_t, nullptr, BT, 128, CDIM);
  // maa einsum as 4 strided [BT,32]@[32,C] GEMMs
  float* mouts[4] = {mrg, mwa, mk_m, mv_m};
  for (int f = 0; f < 4; ++f)
    gemm_bf16<float, EPI_NONE, false><<<gg(BT, CDIM), 256, 0, stream>>>(
        m_t + f * 32, 128, wb_maa_w2 + (size_t)f * 32 * CDIM, CDIM,
        mouts[f], nullptr, BT, CDIM, 32);
  mix4_kernel<<<eltGrid, eltBlk, 0, stream>>>(
      xln, xx, mrg, mwa, mk_m, mv_m, maa_rg, maa_wa, maa_k, maa_v,
      xrg_b, xwa_b, xk_b, xv_b);

  float* rbuf = mrg;   // m buffers consumed by mix4; reuse slots
  float* wlog = mwa;
  float* kbuf = mk_m;
  float* vbuf = mv_m;
  gemm_bf16<float, EPI_NONE, false><<<gg(BT, CDIM), 256, 0, stream>>>(
      xrg_b, CDIM, wb_r, CDIM, rbuf, nullptr, BT, CDIM, CDIM);
  gemm_bf16<float, EPI_NONE, false><<<gg(BT, CDIM), 256, 0, stream>>>(
      xk_b, CDIM, wb_k, CDIM, kbuf, nullptr, BT, CDIM, CDIM);
  gemm_bf16<float, EPI_NONE, false><<<gg(BT, CDIM), 256, 0, stream>>>(
      xv_b, CDIM, wb_v, CDIM, vbuf, nullptr, BT, CDIM, CDIM);
  // gate
  gemm_bf16<__bf16, EPI_TANH, false><<<gg(BT, 128), 256, 0, stream>>>(
      xrg_b, CDIM, wb_gate_w1, 128, gate_t, nullptr, BT, 128, CDIM);
  gemm_bf16<float, EPI_NONE, false><<<gg(BT, CDIM), 256, 0, stream>>>(
      gate_t, 128, wb_gate_w2, CDIM, gbuf, nullptr, BT, CDIM, 128);
  // decay
  gemm_bf16<__bf16, EPI_TANH, false><<<gg(BT, 64), 256, 0, stream>>>(
      xwa_b, CDIM, wb_dec_w1, 64, dec_t, nullptr, BT, 64, CDIM);
  gemm_bf16<float, EPI_NONE, false><<<gg(BT, CDIM), 256, 0, stream>>>(
      dec_t, 64, wb_dec_w2, CDIM, wlog, nullptr, BT, CDIM, 64);
  decay_kernel<<<eltGrid, eltBlk, 0, stream>>>(wlog, time_dec);
  // a / ma / mk2 bottlenecks (K padded to 32; sigmoid+bias fused)
  gemm_bf16<__bf16, EPI_NONE, false><<<gg(BT, 32), 256, 0, stream>>>(
      xwa_b, CDIM, wb_aaa_w1, 32, aaa_t, nullptr, BT, 32, CDIM);
  gemm_bf16<float, EPI_SIGB, false><<<gg(BT, CDIM), 256, 0, stream>>>(
      aaa_t, 32, wb_aaa_w2, CDIM, asig, aaaaa, BT, CDIM, 32);
  gemm_bf16<__bf16, EPI_NONE, false><<<gg(BT, 32), 256, 0, stream>>>(
      xwa_b, CDIM, wb_ma_w1, 32, ma_t, nullptr, BT, 32, CDIM);
  gemm_bf16<float, EPI_SIGB, false><<<gg(BT, CDIM), 256, 0, stream>>>(
      ma_t, 32, wb_ma_w2, CDIM, mabuf, misc_a, BT, CDIM, 32);
  gemm_bf16<__bf16, EPI_NONE, false><<<gg(BT, 32), 256, 0, stream>>>(
      xk_b, CDIM, wb_mk_w1, 32, mk_t, nullptr, BT, 32, CDIM);
  gemm_bf16<float, EPI_SIGB, false><<<gg(BT, CDIM), 256, 0, stream>>>(
      mk_t, 32, wb_mk_w2, CDIM, mk2, misc_k, BT, CDIM, 32);
  // kk (from unmodified k)
  gemm_bf16<__bf16, EPI_TANH, false><<<gg(BT, 32), 256, 0, stream>>>(
      xk_b, CDIM, wb_kkk_w1, 32, kkk_t, nullptr, BT, 32, CDIM);
  gemm_bf16<float, EPI_NONE, false><<<gg(BT, CDIM), 256, 0, stream>>>(
      kkk_t, 32, wb_kkk_w2, CDIM, kkraw, nullptr, BT, CDIM, 32);
  kknorm_kernel<<<BT * NH, 64, 0, stream>>>(kbuf, kkraw, kkbuf);
  kmod_kernel<<<eltGrid, eltBlk, 0, stream>>>(kbuf, mabuf, asig, wlog, mk2);

  // sequential state scan
  wkv7_kernel<<<BB * NH, 256, 0, stream>>>(rbuf, wlog, kbuf, vbuf, kkbuf, asig,
                                           obuf);

  gngate_kernel<<<BT * NH, 64, 0, stream>>>(obuf, rbuf, kbuf, vbuf, gbuf,
                                            faaaa, gn_w, gn_b, og_b);
  gemm_bf16<float, EPI_NONE, false><<<gg(BT, CDIM), 256, 0, stream>>>(
      og_b, CDIM, wb_o, CDIM, oproj, nullptr, BT, CDIM, CDIM);
  add_kernel<<<eltGrid, eltBlk, 0, stream>>>(x, oproj, out, NELT);  // h

  // ---- cmix --------------------------------------------------------------
  float* hln = xln;
  ln_kernel<<<BT, 256, 0, stream>>>(out, ln2_w, ln2_b, hln);
  shiftcoef_kernel<<<eltGrid, eltBlk, 0, stream>>>(hln, cm_maa_k, kf_b);
  gemm_bf16<__bf16, EPI_RELU2, false><<<gg(BT, FFND), 256, 0, stream>>>(
      kf_b, CDIM, wb_cmkey, FFND, ffb, nullptr, BT, FFND, CDIM);
  // d_out += relu(kf@cm_key)^2 @ cm_val
  gemm_bf16<float, EPI_NONE, true><<<gg(BT, CDIM), 256, 0, stream>>>(
      ffb, FFND, wb_cmval, CDIM, out, nullptr, BT, CDIM, FFND);
}
